// LightGQA_65369402245499
// MI455X (gfx1250) — matrix operand order repaired and checked by reading, not compiled, not run on hardware
//
#include <hip/hip_runtime.h>

// ---------------------------------------------------------------------------
// LightGQA for MI455X (gfx1250): bf16 WMMA everywhere dense.
// B=8, L=1024, D=1024, H=16, G=4, HD=64, DK=256, CG=8, K=3
// ---------------------------------------------------------------------------

typedef __bf16 bf16_t;
typedef __attribute__((ext_vector_type(16))) __bf16 v16bf;
typedef __attribute__((ext_vector_type(8)))  __bf16 v8bf;
typedef __attribute__((ext_vector_type(8)))  float  v8f;

__device__ __forceinline__ v8f vzero8() {
    v8f z = {0.f, 0.f, 0.f, 0.f, 0.f, 0.f, 0.f, 0.f};
    return z;
}

// A-operand fragment (16x32 bf16) from a row-major LDS tile [row][k].
// ISA layout: lane m = lane&15, half = lane>>4;
// VGPR 0..3 hold K = half*8 + {0..7}, VGPR 4..7 hold K = 16 + half*8 + {0..7}.
// -> two contiguous 16B chunks per lane (2x ds_load_b128).
__device__ __forceinline__ v16bf ld_frag_a(const bf16_t* base, int stride) {
    const int lane = threadIdx.x & 31;
    const bf16_t* p = base + (lane & 15) * stride + (lane >> 4) * 8;
    union { v16bf v; v8bf h[2]; } u;
    u.h[0] = *(const v8bf*)(p);
    u.h[1] = *(const v8bf*)(p + 16);
    return u.v;
}

// B-operand fragment (32x16 bf16) from a row-major LDS tile [n][k]:
// lane n = lane&15, half = lane>>4; K = half*8 + {0..7}, then 16 + half*8 + {0..7}.
__device__ __forceinline__ v16bf ld_frag_b(const bf16_t* base, int stride) {
    const int lane = threadIdx.x & 31;
    const bf16_t* p = base + (lane & 15) * stride + (lane >> 4) * 8;
    union { v16bf v; v8bf h[2]; } u;
    u.h[0] = *(const v8bf*)(p);
    u.h[1] = *(const v8bf*)(p + 16);
    return u.v;
}

__device__ __forceinline__ v8f wmma_bf16(v16bf a, v16bf b, v8f c) {
    return __builtin_amdgcn_wmma_f32_16x16x32_bf16(
        /*neg_a=*/false, a, /*neg_b=*/false, b,
        /*c_mod=*/(short)0, c, /*reuse_a=*/false, /*reuse_b=*/false);
}

// ---------------------------------------------------------------------------
// Generic C = A[f32 MxK] * B[f32 KxN] + bias, bf16 WMMA, 128x128 tiles,
// 8 waves (4 row-groups x 2 col-groups), double-buffered LDS.
// ---------------------------------------------------------------------------
__global__ __launch_bounds__(256)
void gemm_bf16_kernel(const float* __restrict__ A, const float* __restrict__ Bm,
                      const float* __restrict__ bias, float* __restrict__ C,
                      int M, int N, int K) {
    __shared__ alignas(16) bf16_t As[2][128 * 32];   // [m][k]
    __shared__ alignas(16) bf16_t Bs[2][128 * 32];   // transposed: [n][k]
    const int tid  = threadIdx.x;
    const int lane = tid & 31;
    const int w    = tid >> 5;
    const int rw   = w & 3;      // row wave group (32 rows)
    const int cw   = w >> 2;     // col wave group (64 cols)
    const int m0   = blockIdx.y * 128;
    const int n0   = blockIdx.x * 128;

    v8f acc[2][4];
#pragma unroll
    for (int i = 0; i < 2; ++i)
#pragma unroll
        for (int j = 0; j < 4; ++j) acc[i][j] = vzero8();

    // Batch all global loads of a stage before converting so the 8
    // global_load_b128s pipeline instead of each waiting loadcnt==0.
    auto stage = [&](int kt, int buf) {
        const int k0 = kt * 32;
        float4 xa[4], xb[4];
#pragma unroll
        for (int i = 0; i < 4; ++i) {           // A: 128x32 f32
            int id = tid + i * 256;             // 1024 float4
            xa[i] = *(const float4*)(A + (size_t)(m0 + (id >> 3)) * K + k0 + (id & 7) * 4);
        }
#pragma unroll
        for (int i = 0; i < 4; ++i) {           // B: 32x128 f32
            int id = tid + i * 256;
            xb[i] = *(const float4*)(Bm + (size_t)(k0 + (id >> 5)) * N + n0 + (id & 31) * 4);
        }
#pragma unroll
        for (int i = 0; i < 4; ++i) {
            int id = tid + i * 256;
            int r  = id >> 3;
            int c4 = (id & 7) * 4;
            bf16_t* dst = &As[buf][r * 32 + c4];
            dst[0] = (bf16_t)xa[i].x; dst[1] = (bf16_t)xa[i].y;
            dst[2] = (bf16_t)xa[i].z; dst[3] = (bf16_t)xa[i].w;
        }
#pragma unroll
        for (int i = 0; i < 4; ++i) {
            int id = tid + i * 256;
            int kk = id >> 5;
            int n4 = (id & 31) * 4;
            Bs[buf][(n4 + 0) * 32 + kk] = (bf16_t)xb[i].x;
            Bs[buf][(n4 + 1) * 32 + kk] = (bf16_t)xb[i].y;
            Bs[buf][(n4 + 2) * 32 + kk] = (bf16_t)xb[i].z;
            Bs[buf][(n4 + 3) * 32 + kk] = (bf16_t)xb[i].w;
        }
    };

    const int nk = K >> 5;
    stage(0, 0);
    __syncthreads();
    int buf = 0;
    for (int kt = 0; kt < nk; ++kt) {
        if (kt + 1 < nk) stage(kt + 1, buf ^ 1);
        // Load all fragments into distinct registers, then a dense WMMA clause.
        v16bf a0 = ld_frag_a(&As[buf][(rw * 32 + 0)  * 32], 32);
        v16bf a1 = ld_frag_a(&As[buf][(rw * 32 + 16) * 32], 32);
        v16bf bfr[4];
#pragma unroll
        for (int ct = 0; ct < 4; ++ct)
            bfr[ct] = ld_frag_b(&Bs[buf][(cw * 64 + ct * 16) * 32], 32);
#pragma unroll
        for (int ct = 0; ct < 4; ++ct) {
            acc[0][ct] = wmma_bf16(a0, bfr[ct], acc[0][ct]);
            acc[1][ct] = wmma_bf16(a1, bfr[ct], acc[1][ct]);
        }
        __syncthreads();
        buf ^= 1;
    }

    const int half = lane >> 4, nloc = lane & 15;
#pragma unroll
    for (int rt = 0; rt < 2; ++rt)
#pragma unroll
        for (int ct = 0; ct < 4; ++ct) {
            int n = n0 + cw * 64 + ct * 16 + nloc;
            float bv = bias[n];
#pragma unroll
            for (int v = 0; v < 8; ++v) {
                int m = m0 + rw * 32 + rt * 16 + v + 8 * half;
                C[(size_t)m * N + n] = acc[rt][ct][v] + bv;
            }
        }
}

// ---------------------------------------------------------------------------
// Grouped temporal conv (K=3, pad 1, CG=8 groups of 32 chans) on k & v slices
// of qkv. Writes K as bf16 [b][g][t][d] and V transposed as bf16 [b][g][d][t].
// ---------------------------------------------------------------------------
__global__ __launch_bounds__(256)
void conv_kernel(const float* __restrict__ qkv, const float* __restrict__ cwt,
                 const float* __restrict__ cb, bf16_t* __restrict__ kbf,
                 bf16_t* __restrict__ vtb) {
    __shared__ float xk[3][256];
    __shared__ float xv[3][256];
    const int bl = blockIdx.x;           // b*1024 + l
    const int b = bl >> 10, l = bl & 1023;
    const int co = threadIdx.x;
#pragma unroll
    for (int t = 0; t < 3; ++t) {
        int ll = l + t - 1;
        bool ok = (ll >= 0) && (ll < 1024);
        size_t row = (size_t)(b * 1024 + (ok ? ll : 0)) * 1536;
        xk[t][co] = ok ? qkv[row + 1024 + co] : 0.f;
        xv[t][co] = ok ? qkv[row + 1280 + co] : 0.f;
    }
    __syncthreads();
    const int gi = co >> 5;
    const float* wrow = cwt + co * 96;   // [32][3]
    float yk = cb[co], yv = cb[co];
#pragma unroll
    for (int ci = 0; ci < 32; ++ci) {
        float w0 = wrow[ci * 3 + 0], w1 = wrow[ci * 3 + 1], w2 = wrow[ci * 3 + 2];
        int c = gi * 32 + ci;
        yk += xk[0][c] * w0 + xk[1][c] * w1 + xk[2][c] * w2;
        yv += xv[0][c] * w0 + xv[1][c] * w1 + xv[2][c] * w2;
    }
    const int g = co >> 6, d = co & 63;
    kbf[((size_t)(b * 4 + g) * 1024 + l) * 64 + d] = (bf16_t)yk;
    vtb[((size_t)(b * 4 + g) * 64 + d) * 1024 + l] = (bf16_t)yv;
}

// Shared helper: stage one head's 64x64 Q block (f32 -> bf16) into LDS.
__device__ __forceinline__ void stage_q(const float* __restrict__ qkv,
                                        bf16_t* Qs, int bl0, int h, int tid) {
    float4 xq[4];
#pragma unroll
    for (int i = 0; i < 4; ++i) {
        int id = tid + i * 256;                // 1024 float4
        xq[i] = *(const float4*)(qkv + (size_t)(bl0 + (id >> 4)) * 1536 + h * 64 + (id & 15) * 4);
    }
#pragma unroll
    for (int i = 0; i < 4; ++i) {
        int id = tid + i * 256;
        int r = id >> 4, c4 = (id & 15) * 4;
        bf16_t* dst = &Qs[r * 64 + c4];
        dst[0] = (bf16_t)xq[i].x; dst[1] = (bf16_t)xq[i].y;
        dst[2] = (bf16_t)xq[i].z; dst[3] = (bf16_t)xq[i].w;
    }
}

// ---------------------------------------------------------------------------
// Attention pass 1: per (b, head, 64-row tile) compute softmax row max / sum.
// Scores via WMMA; row reductions via 16-lane butterflies (C-layout halves).
// ---------------------------------------------------------------------------
__global__ __launch_bounds__(256)
void attn_pass1(const float* __restrict__ qkv, const bf16_t* __restrict__ kbf,
                float* __restrict__ Mrow, float* __restrict__ Lrow) {
    __shared__ alignas(16) bf16_t Qs[64 * 64];
    __shared__ alignas(16) bf16_t Ks[32 * 64];
    __shared__ float redm[2][64];
    __shared__ float reds[2][64];
    const int tid = threadIdx.x, lane = tid & 31, w = tid >> 5;
    const int rw = w & 3, cw = w >> 2;
    const int lt = blockIdx.x, h = blockIdx.y, b = blockIdx.z;
    const int g = h & 3;
    const int bl0 = b * 1024 + lt * 64;

    stage_q(qkv, Qs, bl0, h, tid);

    float mrun[8], srun[8];
#pragma unroll
    for (int v = 0; v < 8; ++v) { mrun[v] = -1e30f; srun[v] = 0.f; }
    const bf16_t* kbase = kbf + (size_t)(b * 4 + g) * 1024 * 64;
    const float scale = 0.125f;                // 1/sqrt(64)

    for (int tt = 0; tt < 32; ++tt) {
        __syncthreads();
        {   // stage K tile 32x64 bf16 (one uint4 per thread)
            int r = tid >> 3, c8 = (tid & 7) * 8;
            *(uint4*)(&Ks[r * 64 + c8]) =
                *(const uint4*)(kbase + (size_t)(tt * 32 + r) * 64 + c8);
        }
        __syncthreads();
        v16bf qa0 = ld_frag_a(&Qs[(rw * 16) * 64], 64);
        v16bf qa1 = ld_frag_a(&Qs[(rw * 16) * 64 + 32], 64);
        v16bf kb0 = ld_frag_b(&Ks[(cw * 16) * 64], 64);
        v16bf kb1 = ld_frag_b(&Ks[(cw * 16) * 64 + 32], 64);
        v8f acc = vzero8();
        acc = wmma_bf16(qa0, kb0, acc);
        acc = wmma_bf16(qa1, kb1, acc);
#pragma unroll
        for (int v = 0; v < 8; ++v) {
            float s = acc[v] * scale;
            float tmax = s;
            tmax = fmaxf(tmax, __shfl_xor(tmax, 1, 32));
            tmax = fmaxf(tmax, __shfl_xor(tmax, 2, 32));
            tmax = fmaxf(tmax, __shfl_xor(tmax, 4, 32));
            tmax = fmaxf(tmax, __shfl_xor(tmax, 8, 32));
            float nm = fmaxf(mrun[v], tmax);
            float e = __expf(s - nm);
            e += __shfl_xor(e, 1, 32);
            e += __shfl_xor(e, 2, 32);
            e += __shfl_xor(e, 4, 32);
            e += __shfl_xor(e, 8, 32);
            srun[v] = srun[v] * __expf(mrun[v] - nm) + e;
            mrun[v] = nm;
        }
    }
    if ((lane & 15) == 0) {
        int half = lane >> 4;
#pragma unroll
        for (int v = 0; v < 8; ++v) {
            redm[cw][rw * 16 + half * 8 + v] = mrun[v];
            reds[cw][rw * 16 + half * 8 + v] = srun[v];
        }
    }
    __syncthreads();
    if (tid < 64) {
        float m0 = redm[0][tid], m1 = redm[1][tid];
        float s0 = reds[0][tid], s1 = reds[1][tid];
        float m = fmaxf(m0, m1);
        float s = s0 * __expf(m0 - m) + s1 * __expf(m1 - m);
        size_t idx = (size_t)(b * 16 + h) * 1024 + lt * 64 + tid;
        Mrow[idx] = m;
        Lrow[idx] = s;
    }
}

// ---------------------------------------------------------------------------
// Attention pass 2: per (b, head, 64-row tile) recompute scores, normalize,
// bounce P through LDS (C-layout -> A-layout), O += P*V with WMMA.
// ---------------------------------------------------------------------------
__global__ __launch_bounds__(256)
void attn_out(const float* __restrict__ qkv, const bf16_t* __restrict__ kbf,
              const bf16_t* __restrict__ vtb, const float* __restrict__ Mrow,
              const float* __restrict__ Lrow, float* __restrict__ outpre) {
    __shared__ alignas(16) bf16_t Qs[64 * 64];
    __shared__ alignas(16) bf16_t Ks[32 * 64];
    __shared__ alignas(16) bf16_t Vts[64 * 32];   // [d][t]
    __shared__ alignas(16) bf16_t Ps[64 * 32];    // [l][t]
    const int tid = threadIdx.x, lane = tid & 31, w = tid >> 5;
    const int rw = w & 3, cw = w >> 2;
    const int lt = blockIdx.x, h = blockIdx.y, b = blockIdx.z;
    const int g = h & 3;
    const int bl0 = b * 1024 + lt * 64;
    const int half = lane >> 4, nloc = lane & 15;

    stage_q(qkv, Qs, bl0, h, tid);

    float Mreg[8], Lrec[8];
    {
        size_t sbase = (size_t)(b * 16 + h) * 1024 + lt * 64 + rw * 16 + half * 8;
#pragma unroll
        for (int v = 0; v < 8; ++v) {
            Mreg[v] = Mrow[sbase + v];
            Lrec[v] = 1.f / Lrow[sbase + v];
        }
    }
    v8f o0 = vzero8(), o1 = vzero8();
    const bf16_t* kbase = kbf + (size_t)(b * 4 + g) * 1024 * 64;
    const bf16_t* vbase = vtb + (size_t)(b * 4 + g) * 64 * 1024;

    for (int tt = 0; tt < 32; ++tt) {
        __syncthreads();
        {
            int r = tid >> 3, c8 = (tid & 7) * 8;
            *(uint4*)(&Ks[r * 64 + c8]) =
                *(const uint4*)(kbase + (size_t)(tt * 32 + r) * 64 + c8);
        }
        {
            int dr = tid >> 2, t8 = (tid & 3) * 8;
            *(uint4*)(&Vts[dr * 32 + t8]) =
                *(const uint4*)(vbase + (size_t)dr * 1024 + tt * 32 + t8);
        }
        __syncthreads();
        v16bf qa0 = ld_frag_a(&Qs[(rw * 16) * 64], 64);
        v16bf qa1 = ld_frag_a(&Qs[(rw * 16) * 64 + 32], 64);
        v16bf kb0 = ld_frag_b(&Ks[(cw * 16) * 64], 64);
        v16bf kb1 = ld_frag_b(&Ks[(cw * 16) * 64 + 32], 64);
        v8f acc = vzero8();
        acc = wmma_bf16(qa0, kb0, acc);
        acc = wmma_bf16(qa1, kb1, acc);
#pragma unroll
        for (int v = 0; v < 8; ++v) {
            float p = __expf(acc[v] * 0.125f - Mreg[v]) * Lrec[v];
            int row = rw * 16 + half * 8 + v;
            Ps[row * 32 + cw * 16 + nloc] = (bf16_t)p;
        }
        __syncthreads();
        v16bf pa = ld_frag_a(&Ps[(rw * 16) * 32], 32);
        v16bf vb0 = ld_frag_b(&Vts[(cw * 32) * 32], 32);
        v16bf vb1 = ld_frag_b(&Vts[(cw * 32 + 16) * 32], 32);
        o0 = wmma_bf16(pa, vb0, o0);
        o1 = wmma_bf16(pa, vb1, o1);
    }
#pragma unroll
    for (int v = 0; v < 8; ++v) {
        int row = bl0 + rw * 16 + half * 8 + v;
        outpre[(size_t)row * 1024 + h * 64 + cw * 32 + nloc]      = o0[v];
        outpre[(size_t)row * 1024 + h * 64 + cw * 32 + 16 + nloc] = o1[v];
    }
}

// ---------------------------------------------------------------------------
// attn_mean: per (b, 64-row tile, 128-col t-chunk) loop all 16 heads,
// recompute+normalize scores, accumulate mean in registers. No atomics.
// ---------------------------------------------------------------------------
__global__ __launch_bounds__(256)
void attn_mean(const float* __restrict__ qkv, const bf16_t* __restrict__ kbf,
               const float* __restrict__ Mrow, const float* __restrict__ Lrow,
               float* __restrict__ am) {
    __shared__ alignas(16) bf16_t Qs[64 * 64];
    __shared__ alignas(16) bf16_t Ks4[4][32 * 64];
    const int tid = threadIdx.x, lane = tid & 31, w = tid >> 5;
    const int rw = w & 3, cw = w >> 2;
    const int tc = blockIdx.x, lt = blockIdx.y, b = blockIdx.z;
    const int bl0 = b * 1024 + lt * 64;
    const int half = lane >> 4, nloc = lane & 15;

    float am_acc[4][8];
#pragma unroll
    for (int i = 0; i < 4; ++i)
#pragma unroll
        for (int v = 0; v < 8; ++v) am_acc[i][v] = 0.f;

    for (int g = 0; g < 4; ++g) {
        __syncthreads();
        {   // stage 4 K tiles (t-chunk) for this group: 8192 bf16
            const bf16_t* kbase = kbf + (size_t)(b * 4 + g) * 1024 * 64;
#pragma unroll
            for (int j = 0; j < 4; ++j) {
                int id = tid + j * 256;
                int tile = id >> 8, within = id & 255;
                int r = within >> 3, c8 = (within & 7) * 8;
                *(uint4*)(&Ks4[tile][r * 64 + c8]) =
                    *(const uint4*)(kbase + (size_t)((tc * 4 + tile) * 32 + r) * 64 + c8);
            }
        }
        for (int m = 0; m < 4; ++m) {
            const int h = m * 4 + g;               // head = m*G + g
            __syncthreads();
            stage_q(qkv, Qs, bl0, h, tid);
            __syncthreads();
            float Mreg[8], Lrec[8];
            size_t sbase = (size_t)(b * 16 + h) * 1024 + lt * 64 + rw * 16 + half * 8;
#pragma unroll
            for (int v = 0; v < 8; ++v) {
                Mreg[v] = Mrow[sbase + v];
                Lrec[v] = 1.f / Lrow[sbase + v];
            }
            v16bf qa0 = ld_frag_a(&Qs[(rw * 16) * 64], 64);
            v16bf qa1 = ld_frag_a(&Qs[(rw * 16) * 64 + 32], 64);
#pragma unroll
            for (int tti = 0; tti < 4; ++tti) {
                v16bf kb0 = ld_frag_b(&Ks4[tti][(cw * 16) * 64], 64);
                v16bf kb1 = ld_frag_b(&Ks4[tti][(cw * 16) * 64 + 32], 64);
                v8f acc = vzero8();
                acc = wmma_bf16(qa0, kb0, acc);
                acc = wmma_bf16(qa1, kb1, acc);
#pragma unroll
                for (int v = 0; v < 8; ++v)
                    am_acc[tti][v] += __expf(acc[v] * 0.125f - Mreg[v]) * Lrec[v];
            }
        }
    }
#pragma unroll
    for (int tti = 0; tti < 4; ++tti) {
#pragma unroll
        for (int v = 0; v < 8; ++v) {
            int row  = lt * 64 + rw * 16 + half * 8 + v;
            int tcol = tc * 128 + tti * 32 + cw * 16 + nloc;
            am[((size_t)b * 1024 + row) * 1024 + tcol] = am_acc[tti][v] * 0.0625f;
        }
    }
}

// ---------------------------------------------------------------------------
extern "C" void kernel_launch(void* const* d_in, const int* in_sizes, int n_in,
                              void* d_out, int out_size, void* d_ws, size_t ws_size,
                              hipStream_t stream) {
    (void)in_sizes; (void)n_in; (void)out_size; (void)ws_size;
    const float* query = (const float*)d_in[0];
    // d_in[1]=key, d_in[2]=value: unused by the reference computation.
    const float* Wqkv  = (const float*)d_in[3];
    const float* bqkv  = (const float*)d_in[4];
    const float* convw = (const float*)d_in[5];
    const float* convb = (const float*)d_in[6];
    const float* Wout  = (const float*)d_in[7];
    const float* bout  = (const float*)d_in[8];

    float* out = (float*)d_out;                       // [8192][1024]
    float* am  = out + (size_t)8192 * 1024;           // [8][1024][1024]

    char* ws = (char*)d_ws;
    float*  qkv    = (float*)ws;  ws += (size_t)8192 * 1536 * 4;   // 48 MB
    float*  outpre = (float*)ws;  ws += (size_t)8192 * 1024 * 4;   // 32 MB
    bf16_t* kbf    = (bf16_t*)ws; ws += (size_t)8 * 4 * 1024 * 64 * 2;  // 4 MB
    bf16_t* vtb    = (bf16_t*)ws; ws += (size_t)8 * 4 * 1024 * 64 * 2;  // 4 MB
    float*  Mr     = (float*)ws;  ws += (size_t)8 * 16 * 1024 * 4;
    float*  Lr     = (float*)ws;  ws += (size_t)8 * 16 * 1024 * 4;

    gemm_bf16_kernel<<<dim3(1536 / 128, 8192 / 128), 256, 0, stream>>>(
        query, Wqkv, bqkv, qkv, 8192, 1536, 1024);
    conv_kernel<<<8192, 256, 0, stream>>>(qkv, convw, convb, kbf, vtb);
    attn_pass1<<<dim3(16, 16, 8), 256, 0, stream>>>(qkv, kbf, Mr, Lr);
    attn_out<<<dim3(16, 16, 8), 256, 0, stream>>>(qkv, kbf, vtb, Mr, Lr, outpre);
    attn_mean<<<dim3(8, 16, 8), 256, 0, stream>>>(qkv, kbf, Mr, Lr, am);
    gemm_bf16_kernel<<<dim3(1024 / 128, 8192 / 128), 256, 0, stream>>>(
        outpre, Wout, bout, out, 8192, 1024, 1024);
}
